// GroupedQueryAttention_2525440770388
// MI455X (gfx1250) — matrix operand order repaired and checked by reading, not compiled, not run on hardware
//
#include <hip/hip_runtime.h>

typedef __attribute__((ext_vector_type(16))) __bf16 v16bf;
typedef __attribute__((ext_vector_type(8)))  float  v8f;
typedef __attribute__((ext_vector_type(4)))  unsigned int u32x4;
typedef __attribute__((ext_vector_type(4)))  int i32x4;
typedef __attribute__((ext_vector_type(8)))  int i32x8;

union Frag16 {
  unsigned short u[16];
  u32x4 q[2];
  v16bf v;
};

constexpr int SEQ  = 2048;
constexpr int DIN  = 2048;
constexpr int NH   = 32;
constexpr int NKV  = 8;
constexpr int HDIM = 64;
constexpr int DOUT = NH * HDIM;   // 2048
constexpr int KVD  = NKV * HDIM;  // 512

#if __has_builtin(__builtin_amdgcn_tensor_load_to_lds)
#define HAVE_TDM 1
#else
#define HAVE_TDM 0
#endif

__device__ __forceinline__ unsigned short f2bf(float f) {
  unsigned int u = __builtin_bit_cast(unsigned int, f);
  u += 0x7FFFu + ((u >> 16) & 1u);          // round-to-nearest-even
  return (unsigned short)(u >> 16);
}

__device__ __forceinline__ unsigned ldsoff(const void* p) {
  return (unsigned)(unsigned long long)p;   // low 32 bits of generic addr = LDS offset
}

// CDNA5 async global->LDS copy, 16B per enabled lane (ASYNCcnt path).
__device__ __forceinline__ void async_g2l_b128(unsigned lds, const void* gp) {
  asm volatile("global_load_async_to_lds_b128 %0, %1, off"
               :: "v"(lds), "v"(gp) : "memory");
}
__device__ __forceinline__ void wait_async0() {
  asm volatile("s_wait_asynccnt 0x0" ::: "memory");
}

#if HAVE_TDM
// Tensor Data Mover: 2D tile (tile_d1 rows x tile_d0 contiguous elems, 2B each)
// from a [tensor_d1][tensor_d0-stride] bf16 tensor into LDS, with row padding
// (pad_amount DWORD-code after each pad_interval DWORD-code) so it lands in a
// padded LDS tile. Descriptor per ISA 8.3/8.4; issued by one wave, TENSORcnt.
__device__ __forceinline__ void tdm_load_2d(unsigned lds, const void* g,
                                            unsigned tensor_d0, unsigned tensor_d1,
                                            unsigned tile_d0, unsigned tile_d1,
                                            unsigned stride_d0,
                                            unsigned pad_int_code, unsigned pad_amt_code) {
  unsigned long long ga = (unsigned long long)g;
  u32x4 g0 = {};
  g0[0] = 1u;                                               // count=1, user desc
  g0[1] = lds;                                              // LDS byte address
  g0[2] = (unsigned)ga;                                     // global_addr[31:0]
  g0[3] = ((unsigned)(ga >> 32) & 0x01FFFFFFu) | (2u << 30); // addr[56:32] | type=2
  i32x8 g1 = {};
  g1[0] = (int)((1u << 16)                                  // data_size = 2 bytes
              | (1u << 20)                                  // pad_enable
              | (pad_int_code << 22)                        // pad interval (DWORDs)
              | (pad_amt_code << 25));                      // pad amount (DWORDs)
  g1[1] = (int)(tensor_d0 << 16);                           // tensor_dim0[15:0]
  g1[2] = (int)((tensor_d0 >> 16) | (tensor_d1 << 16));     // dim0 hi | dim1 lo
  g1[3] = (int)((tensor_d1 >> 16) | (tile_d0 << 16));       // dim1 hi | tile_dim0
  g1[4] = (int)(tile_d1 & 0xFFFFu);                         // tile_dim1 (tile_dim2=0)
  g1[5] = (int)stride_d0;                                   // tensor_dim0_stride lo
  i32x4 z4 = {};
#if defined(__clang_major__) && __clang_major__ >= 23
  i32x8 z8 = {};
  __builtin_amdgcn_tensor_load_to_lds(g0, g1, z4, z4, z8, 0);
#else
  __builtin_amdgcn_tensor_load_to_lds(g0, g1, z4, z4, 0);
#endif
}
#endif

// ---------------------------------------------------------------------------
// fp32 -> bf16, vectorized
// ---------------------------------------------------------------------------
__global__ __launch_bounds__(256) void cvt_f32_to_bf16_v4(const float4* __restrict__ s,
                                                          uint2* __restrict__ d, int n4) {
  for (int i = blockIdx.x * 256 + threadIdx.x; i < n4; i += gridDim.x * 256) {
    float4 f = s[i];
    uint2 o;
    o.x = (unsigned)f2bf(f.x) | ((unsigned)f2bf(f.y) << 16);
    o.y = (unsigned)f2bf(f.z) | ((unsigned)f2bf(f.w) << 16);
    d[i] = o;
  }
}

// ---------------------------------------------------------------------------
// Transpose + convert: src f32 [R][C] -> dst bf16 [C][R]. 32x32 LDS tile.
// ---------------------------------------------------------------------------
__global__ __launch_bounds__(256) void cvt_transpose_bf16(const float* __restrict__ src,
                                                          unsigned short* __restrict__ dst,
                                                          int R, int C) {
  __shared__ float t[32][33];
  const int c0 = blockIdx.x * 32, r0 = blockIdx.y * 32;
  const int tx = threadIdx.x, ty = threadIdx.y;   // block (32,8)
  #pragma unroll
  for (int i = ty; i < 32; i += 8)
    t[i][tx] = src[(size_t)(r0 + i) * C + c0 + tx];
  __syncthreads();
  #pragma unroll
  for (int i = ty; i < 32; i += 8)
    dst[(size_t)(c0 + i) * R + r0 + tx] = f2bf(t[tx][i]);
}

// ---------------------------------------------------------------------------
// C[M,N] f32 = A[M,K] bf16 (row-major) @ BT[N,K] bf16 (B pre-transposed).
// Block tile 64x128, 8 waves (2x4), each wave a 32x32 tile = 4 WMMA accums.
// Double-buffered LDS; staging via async b128 (loads for t+1 overlap WMMAs of t).
// ---------------------------------------------------------------------------
__global__ __launch_bounds__(256) void gemm_bf16_wmma(const unsigned short* __restrict__ A,
                                                      const unsigned short* __restrict__ BT,
                                                      float* __restrict__ C,
                                                      int M, int N, int K) {
  __shared__ __align__(16) unsigned short As[2][64][40];    // [buf][m][k]
  __shared__ __align__(16) unsigned short Bs[2][128][40];   // [buf][n][k]
  const int tid = threadIdx.x;
  const int wid = tid >> 5, lane = tid & 31;
  const int wm = wid >> 2, wn = wid & 3;                    // 2 x 4 wave grid
  const int l15 = lane & 15, lh = (lane >> 4) & 1;
  const int bM = blockIdx.y * 64, bN = blockIdx.x * 128;
  const int sr = tid >> 2, sc = (tid & 3) * 8;              // 16B chunk coords

  auto stage = [&](int kk, int buf) {
    async_g2l_b128(ldsoff(&As[buf][sr][sc]),      A  + (size_t)(bM + sr) * K + kk + sc);
    async_g2l_b128(ldsoff(&Bs[buf][sr][sc]),      BT + (size_t)(bN + sr) * K + kk + sc);
    async_g2l_b128(ldsoff(&Bs[buf][64 + sr][sc]), BT + (size_t)(bN + 64 + sr) * K + kk + sc);
  };

  v8f acc[2][2];
  const v8f zero = {};
  #pragma unroll
  for (int i = 0; i < 2; i++)
    #pragma unroll
    for (int j = 0; j < 2; j++) acc[i][j] = zero;

  const int ak = lh ? 8 : 0;
  const int bk = lh ? 8 : 0;

  stage(0, 0);
  const int T = K / 32;
  for (int t = 0; t < T; ++t) {
    const int buf = t & 1;
    wait_async0();
    __syncthreads();                       // tile t visible; prev readers done
    if (t + 1 < T) stage((t + 1) * 32, buf ^ 1);

    Frag16 a[2], b[2];
    #pragma unroll
    for (int i = 0; i < 2; i++) {
      const int ar = wm * 32 + i * 16 + l15;
      a[i].q[0] = *(const u32x4*)&As[buf][ar][ak];
      a[i].q[1] = *(const u32x4*)&As[buf][ar][16 + ak];
    }
    #pragma unroll
    for (int j = 0; j < 2; j++) {
      const int bc = wn * 32 + j * 16 + l15;
      b[j].q[0] = *(const u32x4*)&Bs[buf][bc][bk];
      b[j].q[1] = *(const u32x4*)&Bs[buf][bc][16 + bk];
    }
    #pragma unroll
    for (int i = 0; i < 2; i++)
      #pragma unroll
      for (int j = 0; j < 2; j++)
        acc[i][j] = __builtin_amdgcn_wmma_f32_16x16x32_bf16(false, a[i].v, false, b[j].v,
                                                            (short)0, acc[i][j], false, false);
    __syncthreads();
  }

  #pragma unroll
  for (int i = 0; i < 2; i++) {
    const int r0 = bM + wm * 32 + i * 16 + lh * 8;
    #pragma unroll
    for (int j = 0; j < 2; j++) {
      const int col = bN + wn * 32 + j * 16 + l15;
      #pragma unroll
      for (int r = 0; r < 8; r++)
        C[(size_t)(r0 + r) * N + col] = acc[i][j][r];
    }
  }
}

// ---------------------------------------------------------------------------
// RMSNorm + RoPE epilogue for Q/K -> bf16 [nh][S][64];
// for V (doNormRope=0): convert + transpose -> bf16 [nh*64][S].
// ---------------------------------------------------------------------------
__global__ __launch_bounds__(64) void qkv_post(const float* __restrict__ in,
                                               const float* __restrict__ nw,
                                               const float* __restrict__ cs,
                                               const float* __restrict__ sn,
                                               unsigned short* __restrict__ out,
                                               int nh, int doNormRope) {
  __shared__ float red[64];
  __shared__ float nv[64];
  const int s  = blockIdx.x / nh;
  const int hh = blockIdx.x % nh;
  const int d  = threadIdx.x;
  float x = in[((size_t)s * nh + hh) * HDIM + d];
  if (doNormRope) {
    red[d] = x * x;
    __syncthreads();
    #pragma unroll
    for (int off = 32; off > 0; off >>= 1) {
      if (d < off) red[d] += red[d + off];
      __syncthreads();
    }
    const float var = red[0] * (1.0f / HDIM);
    const float nx  = x * rsqrtf(var + 1e-6f) * nw[d];
    nv[d] = nx;
    __syncthreads();
    const float rot = (d < 32) ? -nv[d + 32] : nv[d - 32];
    const float res = nx * cs[(size_t)s * HDIM + d] + rot * sn[(size_t)s * HDIM + d];
    out[((size_t)hh * SEQ + s) * HDIM + d] = f2bf(res);
  } else {
    out[((size_t)hh * HDIM + d) * SEQ + s] = f2bf(x);   // V -> [head][dim][seq]
  }
}

// ---------------------------------------------------------------------------
// Flash attention. Block = (head, 128-row q tile); 8 waves x 16 q rows.
// Q: [H][S][64] bf16, K: [KV][S][64] bf16, VT: [KV*64][S] bf16.
// K/V tiles staged with the Tensor Data Mover (TENSORcnt) directly into the
// padded LDS layout (hardware row padding); async-b128 fallback if no builtin.
// ---------------------------------------------------------------------------
__global__ __launch_bounds__(256) void attn_wmma(const unsigned short* __restrict__ Q,
                                                 const unsigned short* __restrict__ Kt,
                                                 const unsigned short* __restrict__ VT,
                                                 unsigned short* __restrict__ ctx) {
  __shared__ __align__(16) unsigned short Ks[32][72];   // [key][dim], 128B data + 16B pad
  __shared__ __align__(16) unsigned short Vs[64][40];   // [dim][key], 64B data + 16B pad
  __shared__ __align__(16) unsigned short Ps[8][16][40];
  const int h   = blockIdx.x >> 4;
  const int qb  = blockIdx.x & 15;
  const int kvh = h >> 2;                               // GS = 4
  const int tid = threadIdx.x;
  const int wid = tid >> 5, lane = tid & 31;
  const int l15 = lane & 15, lh = (lane >> 4) & 1;
  const int qbase = qb * 128 + wid * 16;
  const int ak = lh ? 8 : 0;

  // Q fragments: two 16x32 A-fragments, b128 global loads.
  Frag16 qa[2];
  {
    const u32x4* qp = (const u32x4*)(Q + ((size_t)h * SEQ + qbase + l15) * HDIM);
    #pragma unroll
    for (int dh = 0; dh < 2; dh++) {
      qa[dh].q[0] = qp[dh * 4 + (ak >> 3)];
      qa[dh].q[1] = qp[dh * 4 + 2 + (ak >> 3)];
    }
  }

  const v8f zero = {};
  v8f o[4];
  #pragma unroll
  for (int t = 0; t < 4; t++) o[t] = zero;
  float mR[8], lR[8];
  #pragma unroll
  for (int r = 0; r < 8; r++) { mR[r] = -1e30f; lR[r] = 0.0f; }

  const int kend = qb * 128 + 128;
  const float scale = 0.125f;                           // 1/sqrt(64)

#if !HAVE_TDM
  const int kr = tid >> 3, kc = (tid & 7) * 8;
  const int vr = tid >> 2, vc = (tid & 3) * 8;
#endif

  for (int kb0 = 0; kb0 < kend; kb0 += 32) {
#if HAVE_TDM
    if (wid == 0) {
      // K tile: 32 rows x 64 elems (128B = 32 DW rows, pad 4 DW -> code 4 / 3)
      tdm_load_2d(ldsoff(&Ks[0][0]), Kt + ((size_t)kvh * SEQ + kb0) * HDIM,
                  HDIM, SEQ, HDIM, 32, HDIM, 4u, 3u);
      // V tile: 64 rows x 32 elems (64B = 16 DW rows, pad 4 DW -> code 3 / 3)
      tdm_load_2d(ldsoff(&Vs[0][0]), VT + (size_t)kvh * HDIM * SEQ + kb0,
                  SEQ, HDIM, 32, HDIM, SEQ, 3u, 3u);
      __builtin_amdgcn_s_wait_tensorcnt(0);
    }
    __syncthreads();
#else
    async_g2l_b128(ldsoff(&Ks[kr][kc]), Kt + ((size_t)kvh * SEQ + kb0 + kr) * HDIM + kc);
    async_g2l_b128(ldsoff(&Vs[vr][vc]), VT + ((size_t)kvh * HDIM + vr) * SEQ + kb0 + vc);
    wait_async0();
    __syncthreads();
#endif

    // Scores: S = Q(16x64) * K^T(64x32) as two 16-key tiles.
    v8f sc[2];
    #pragma unroll
    for (int nt = 0; nt < 2; nt++) {
      v8f a = zero;
      #pragma unroll
      for (int dh = 0; dh < 2; dh++) {
        Frag16 kf;
        const int off = dh * 32 + (lh ? 8 : 0);
        kf.q[0] = *(const u32x4*)&Ks[nt * 16 + l15][off];
        kf.q[1] = *(const u32x4*)&Ks[nt * 16 + l15][off + 16];
        a = __builtin_amdgcn_wmma_f32_16x16x32_bf16(false, qa[dh].v, false, kf.v, (short)0, a, false, false);
      }
      sc[nt] = a;
    }

    // Online softmax: each C row lives in one reg across a half-wave.
    #pragma unroll
    for (int r = 0; r < 8; r++) {
      const int qrow = qbase + r + 8 * lh;
      float v0 = sc[0][r] * scale; if (kb0 + l15 > qrow)      v0 = -1e30f;
      float v1 = sc[1][r] * scale; if (kb0 + 16 + l15 > qrow) v1 = -1e30f;
      float mx = fmaxf(v0, v1);
      #pragma unroll
      for (int d = 1; d < 16; d <<= 1) mx = fmaxf(mx, __shfl_xor(mx, d, 32));
      const float nm   = fmaxf(mR[r], mx);
      const float corr = __expf(mR[r] - nm);
      const float e0 = __expf(v0 - nm), e1 = __expf(v1 - nm);
      float rs = e0 + e1;
      #pragma unroll
      for (int d = 1; d < 16; d <<= 1) rs += __shfl_xor(rs, d, 32);
      lR[r] = lR[r] * corr + rs;
      mR[r] = nm;
      #pragma unroll
      for (int t = 0; t < 4; t++) o[t][r] *= corr;
      Ps[wid][r + 8 * lh][l15]      = f2bf(e0);
      Ps[wid][r + 8 * lh][16 + l15] = f2bf(e1);
    }

    // Reload P in A-fragment layout (same-wave LDS RAW; contiguous b128 reads).
    Frag16 pf;
    pf.q[0] = *(const u32x4*)&Ps[wid][l15][ak];
    pf.q[1] = *(const u32x4*)&Ps[wid][l15][16 + ak];

    // O += P(16x32) * V(32x64): VT layout makes V fragments K-contiguous.
    #pragma unroll
    for (int nt = 0; nt < 4; nt++) {
      Frag16 vf;
      const int ko = lh ? 8 : 0;
      vf.q[0] = *(const u32x4*)&Vs[nt * 16 + l15][ko];
      vf.q[1] = *(const u32x4*)&Vs[nt * 16 + l15][ko + 16];
      o[nt] = __builtin_amdgcn_wmma_f32_16x16x32_bf16(false, pf.v, false, vf.v, (short)0, o[nt], false, false);
    }
    __syncthreads();
  }

  #pragma unroll
  for (int r = 0; r < 8; r++) {
    const int qrow = qbase + r + 8 * lh;
    const float inv = 1.0f / lR[r];
    #pragma unroll
    for (int nt = 0; nt < 4; nt++)
      ctx[(size_t)qrow * DOUT + h * HDIM + nt * 16 + l15] = f2bf(o[nt][r] * inv);
  }
}

// ---------------------------------------------------------------------------
// Host-side launch
// ---------------------------------------------------------------------------
extern "C" void kernel_launch(void* const* d_in, const int* in_sizes, int n_in,
                              void* d_out, int out_size, void* d_ws, size_t ws_size,
                              hipStream_t stream) {
  (void)in_sizes; (void)n_in; (void)out_size; (void)ws_size;
  const float* x    = (const float*)d_in[0];
  // d_in[1] = mask (bool) -- causal mask applied analytically in attn_wmma
  const float* cosT = (const float*)d_in[2];
  const float* sinT = (const float*)d_in[3];
  const float* Wq   = (const float*)d_in[4];
  const float* Wk   = (const float*)d_in[5];
  const float* Wv   = (const float*)d_in[6];
  const float* Wo   = (const float*)d_in[7];
  const float* qnw  = (const float*)d_in[8];
  const float* knw  = (const float*)d_in[9];
  float* out = (float*)d_out;

  size_t off = 0;
  auto alloc = [&](size_t bytes) -> void* {
    void* p = (char*)d_ws + off;
    off += (bytes + 255) & ~(size_t)255;
    return p;
  };
  unsigned short* xb   = (unsigned short*)alloc((size_t)SEQ * DIN * 2);   // [S][DIN]
  unsigned short* wqT  = (unsigned short*)alloc((size_t)DOUT * DIN * 2);  // [DOUT][DIN]
  unsigned short* wkT  = (unsigned short*)alloc((size_t)KVD * DIN * 2);   // [KVD][DIN]
  unsigned short* wvT  = (unsigned short*)alloc((size_t)KVD * DIN * 2);
  unsigned short* woT  = (unsigned short*)alloc((size_t)DIN * DOUT * 2);  // [DIN][DOUT]
  float*          qf   = (float*)alloc((size_t)SEQ * DOUT * 4);
  float*          kf   = (float*)alloc((size_t)SEQ * KVD * 4);
  float*          vf   = (float*)alloc((size_t)SEQ * KVD * 4);
  unsigned short* qb2  = (unsigned short*)alloc((size_t)SEQ * DOUT * 2);  // [H][S][64]
  unsigned short* kb2  = (unsigned short*)alloc((size_t)SEQ * KVD * 2);   // [KV][S][64]
  unsigned short* vbT  = (unsigned short*)alloc((size_t)SEQ * KVD * 2);   // [KV*64][S]
  unsigned short* ctxb = (unsigned short*)alloc((size_t)SEQ * DOUT * 2);  // [S][DOUT]

  // 1) operand conversion: x plain; weights transposed so GEMM B is [N][K]
  cvt_f32_to_bf16_v4<<<1024, 256, 0, stream>>>((const float4*)x, (uint2*)xb, SEQ * DIN / 4);
  cvt_transpose_bf16<<<dim3(DOUT / 32, DIN / 32), dim3(32, 8), 0, stream>>>(Wq, wqT, DIN, DOUT);
  cvt_transpose_bf16<<<dim3(KVD / 32,  DIN / 32), dim3(32, 8), 0, stream>>>(Wk, wkT, DIN, KVD);
  cvt_transpose_bf16<<<dim3(KVD / 32,  DIN / 32), dim3(32, 8), 0, stream>>>(Wv, wvT, DIN, KVD);
  cvt_transpose_bf16<<<dim3(DIN / 32, DOUT / 32), dim3(32, 8), 0, stream>>>(Wo, woT, DOUT, DIN);

  // 2) QKV projections (bf16 WMMA, f32 accumulate), block tile 64x128
  gemm_bf16_wmma<<<dim3(DOUT / 128, SEQ / 64), 256, 0, stream>>>(xb, wqT, qf, SEQ, DOUT, DIN);
  gemm_bf16_wmma<<<dim3(KVD  / 128, SEQ / 64), 256, 0, stream>>>(xb, wkT, kf, SEQ, KVD, DIN);
  gemm_bf16_wmma<<<dim3(KVD  / 128, SEQ / 64), 256, 0, stream>>>(xb, wvT, vf, SEQ, KVD, DIN);

  // 3) RMSNorm + RoPE (q,k); V converted + transposed for the PV WMMA layout
  qkv_post<<<SEQ * NH,  64, 0, stream>>>(qf, qnw, cosT, sinT, qb2, NH,  1);
  qkv_post<<<SEQ * NKV, 64, 0, stream>>>(kf, knw, cosT, sinT, kb2, NKV, 1);
  qkv_post<<<SEQ * NKV, 64, 0, stream>>>(vf, qnw, cosT, sinT, vbT, NKV, 0);

  // 4) Flash attention (causal), K/V staged by the Tensor Data Mover
  attn_wmma<<<NH * (SEQ / 128), 256, 0, stream>>>(qb2, kb2, vbT, ctxb);

  // 5) Output projection -> fp32 result
  gemm_bf16_wmma<<<dim3(DIN / 128, SEQ / 64), 256, 0, stream>>>(ctxb, woT, out, SEQ, DIN, DOUT);
}
